// LMHA_20246475833970
// MI455X (gfx1250) — matrix operand order repaired and checked by reading, not compiled, not run on hardware
//
#include <hip/hip_runtime.h>
#include <hip/hip_bf16.h>

// ---------------------------------------------------------------------------
// Linear multi-head attention (Katharopoulos, non-causal) for MI455X/gfx1250.
// Compute-bound (~292 GFLOP vs ~500MB traffic @23.3TB/s): all matrix ops run
// through v_wmma_f32_16x16x32_bf16 (f32 accumulate).
// Projection GEMM: 256x128x32 block tile, 8 waves (4x2), 64x64 wave tile =
// 16 WMMA per K-step, software-pipelined (global->regs overlapped with WMMA,
// then regs->LDS behind the barrier).
// ---------------------------------------------------------------------------

typedef __bf16 bf16;
typedef __attribute__((ext_vector_type(16))) __bf16 v16bf;
typedef __attribute__((ext_vector_type(8)))  __bf16 v8bf;
typedef __attribute__((ext_vector_type(8)))  float  v8f;
typedef __attribute__((ext_vector_type(4)))  unsigned int u32x4;

#define S_LEN   2048
#define B_SZ    16
#define D_MODEL 1024
#define NH      8
#define DH      128
#define LN10K   9.210340371976184f

__device__ __forceinline__ v16bf cat8(v8bf a, v8bf b) {
  return __builtin_shufflevector(a, b, 0,1,2,3,4,5,6,7,8,9,10,11,12,13,14,15);
}

__global__ __launch_bounds__(256) void k_embed(const float* __restrict__ x,
                                               bf16* __restrict__ h) {
  int i = blockIdx.x * 256 + threadIdx.x;
  int d = i & (D_MODEL - 1);
  int s = i / (B_SZ * D_MODEL);
  float div = __expf((float)(d & ~1) * (-LN10K / (float)D_MODEL));
  float ang = (float)s * div;
  float pe  = (d & 1) ? __cosf(ang) : __sinf(ang);
  h[i] = (bf16)(x[i] + pe);
}

__global__ __launch_bounds__(256) void k_cvt(const float* __restrict__ src,
                                             bf16* __restrict__ dst, int n) {
  int i = blockIdx.x * 256 + threadIdx.x;
  if (i < n) dst[i] = (bf16)src[i];
}

// C[M,N] = A[M,K] @ W[N,K]^T (+bias).  MODE 0: bf16 out, 1: bf16 out + phi
// (elu+1), 2: f32 out.  Block 256x128x32; 8 waves; wave tile 64x64.
template <int MODE>
__global__ __launch_bounds__(256) void k_gemm_xwT(
    const bf16* __restrict__ A, const bf16* __restrict__ W,
    const float* __restrict__ bias, void* __restrict__ Cout,
    int M, int N, int K) {
  __shared__ bf16 sA[256][40];   // [m][k], 80B padded rows (conflict-free b128)
  __shared__ bf16 sB[128][40];   // [n][k]
  const int t = threadIdx.x;
  const int wave = t >> 5, lane = t & 31;
  const int lo = lane & 15, hi = lane >> 4;
  const int wm = wave >> 1, wn = wave & 1;          // 4 waves in M, 2 in N
  const int m0 = blockIdx.x * 256;
  const int n0 = blockIdx.y * 128;

  const v8f vz = {0.f,0.f,0.f,0.f,0.f,0.f,0.f,0.f};
  v8f acc[4][4];
#pragma unroll
  for (int i = 0; i < 4; ++i)
#pragma unroll
    for (int j = 0; j < 4; ++j) acc[i][j] = vz;

  // ---- prologue: stage K-tile 0 straight into LDS ----
#pragma unroll
  for (int i2 = 0; i2 < 4; ++i2) {          // A: 1024 16B chunks, 4/thread
    const int c = t + i2 * 256, row = c >> 2, part = c & 3;
    *reinterpret_cast<u32x4*>(&sA[row][part * 8]) =
        *reinterpret_cast<const u32x4*>(A + (size_t)(m0 + row) * K + part * 8);
  }
#pragma unroll
  for (int i2 = 0; i2 < 2; ++i2) {          // B: 512 16B chunks, 2/thread
    const int c = t + i2 * 256, row = c >> 2, part = c & 3;
    *reinterpret_cast<u32x4*>(&sB[row][part * 8]) =
        *reinterpret_cast<const u32x4*>(W + (size_t)(n0 + row) * K + part * 8);
  }
  __syncthreads();

  for (int k0 = 0; k0 < K; k0 += 32) {
    const bool more = (k0 + 32) < K;
    u32x4 ra[4], rb[2];
    if (more) {                              // prefetch next tile into regs
#pragma unroll
      for (int i2 = 0; i2 < 4; ++i2) {
        const int c = t + i2 * 256, row = c >> 2, part = c & 3;
        ra[i2] = *reinterpret_cast<const u32x4*>(
            A + (size_t)(m0 + row) * K + k0 + 32 + part * 8);
      }
#pragma unroll
      for (int i2 = 0; i2 < 2; ++i2) {
        const int c = t + i2 * 256, row = c >> 2, part = c & 3;
        rb[i2] = *reinterpret_cast<const u32x4*>(
            W + (size_t)(n0 + row) * K + k0 + 32 + part * 8);
      }
    }

    v16bf af[4], bfr[4];
#pragma unroll
    for (int i = 0; i < 4; ++i) {            // A frag: K split 8*hi / 16+8*hi
      const int m = wm * 64 + i * 16 + lo;
      af[i] = cat8(*reinterpret_cast<const v8bf*>(&sA[m][8 * hi]),
                   *reinterpret_cast<const v8bf*>(&sA[m][16 + 8 * hi]));
    }
#pragma unroll
    for (int j = 0; j < 4; ++j) {            // B frag: K split 8*hi / 16+8*hi
      const int n = wn * 64 + j * 16 + lo;
      bfr[j] = cat8(*reinterpret_cast<const v8bf*>(&sB[n][8 * hi]),
                    *reinterpret_cast<const v8bf*>(&sB[n][16 + 8 * hi]));
    }
#pragma unroll
    for (int i = 0; i < 4; ++i)
#pragma unroll
      for (int j = 0; j < 4; ++j)
        acc[i][j] = __builtin_amdgcn_wmma_f32_16x16x32_bf16(
            false, af[i], false, bfr[j], (short)0, acc[i][j], false, false);

    if (more) {
      __syncthreads();                       // everyone done reading this tile
#pragma unroll
      for (int i2 = 0; i2 < 4; ++i2) {
        const int c = t + i2 * 256, row = c >> 2, part = c & 3;
        *reinterpret_cast<u32x4*>(&sA[row][part * 8]) = ra[i2];
      }
#pragma unroll
      for (int i2 = 0; i2 < 2; ++i2) {
        const int c = t + i2 * 256, row = c >> 2, part = c & 3;
        *reinterpret_cast<u32x4*>(&sB[row][part * 8]) = rb[i2];
      }
      __syncthreads();                       // new tile visible
    }
  }

#pragma unroll
  for (int i = 0; i < 4; ++i)
#pragma unroll
    for (int j = 0; j < 4; ++j) {
      const int gn = n0 + wn * 64 + j * 16 + lo;
      const float bv = bias[gn];
#pragma unroll
      for (int r = 0; r < 8; ++r) {
        const int gm = m0 + wm * 64 + i * 16 + 8 * hi + r;
        float v = acc[i][j][r] + bv;
        if (MODE == 1) v = (v > 0.f) ? (v + 1.f) : __expf(v);
        if (MODE == 2)
          reinterpret_cast<float*>(Cout)[(size_t)gm * N + gn] = v;
        else
          reinterpret_cast<bf16*>(Cout)[(size_t)gm * N + gn] = (bf16)v;
      }
    }
}

__global__ __launch_bounds__(256) void k_zsum(const bf16* __restrict__ pk,
                                              float* __restrict__ z) {
  int i = blockIdx.x * 256 + threadIdx.x;
  int b = i / D_MODEL, j = i & (D_MODEL - 1);
  const bf16* p = pk + (size_t)b * D_MODEL + j;
  float s = 0.f;
  for (int sIdx = 0; sIdx < S_LEN; ++sIdx)
    s += (float)p[(size_t)sIdx * B_SZ * D_MODEL];
  z[i] = s;
}

// kv^T[b,h][e,d] = sum_s v[s,b,h,e] * pk[s,b,h,d]; K = s (LDS-transposed).
__global__ __launch_bounds__(256) void k_kv(const bf16* __restrict__ pk,
                                            const bf16* __restrict__ v,
                                            bf16* __restrict__ kvT) {
  __shared__ bf16 sA[128][40];   // [d][s]
  __shared__ bf16 sB[128][40];   // [e][s]
  const int bh = blockIdx.x;
  const int b = bh >> 3, hh = bh & 7;
  const int t = threadIdx.x;
  const int wave = t >> 5, lane = t & 31;
  const int lo = lane & 15, hi = lane >> 4;
  const int wm = wave >> 1, wn = wave & 1;
  const size_t headOff = (size_t)b * D_MODEL + hh * DH;
  const size_t rowStride = (size_t)B_SZ * D_MODEL;

  const v8f vz = {0.f,0.f,0.f,0.f,0.f,0.f,0.f,0.f};
  v8f acc[2][4];
#pragma unroll
  for (int i = 0; i < 2; ++i)
#pragma unroll
    for (int j = 0; j < 4; ++j) acc[i][j] = vz;

  const int s2 = t >> 4;
  const int dg = t & 15;

  for (int s0 = 0; s0 < S_LEN; s0 += 32) {
    const size_t r0 = (size_t)(s0 + 2 * s2) * rowStride + headOff + dg * 8;
    const u32x4 pa0 = *reinterpret_cast<const u32x4*>(pk + r0);
    const u32x4 pa1 = *reinterpret_cast<const u32x4*>(pk + r0 + rowStride);
    const u32x4 pb0 = *reinterpret_cast<const u32x4*>(v + r0);
    const u32x4 pb1 = *reinterpret_cast<const u32x4*>(v + r0 + rowStride);
    const unsigned short* a0 = reinterpret_cast<const unsigned short*>(&pa0);
    const unsigned short* a1 = reinterpret_cast<const unsigned short*>(&pa1);
    const unsigned short* b0 = reinterpret_cast<const unsigned short*>(&pb0);
    const unsigned short* b1 = reinterpret_cast<const unsigned short*>(&pb1);
#pragma unroll
    for (int i = 0; i < 8; ++i) {            // write (s,s+1) pairs as b32
      *reinterpret_cast<unsigned*>(&sA[dg * 8 + i][2 * s2]) =
          (unsigned)a0[i] | ((unsigned)a1[i] << 16);
      *reinterpret_cast<unsigned*>(&sB[dg * 8 + i][2 * s2]) =
          (unsigned)b0[i] | ((unsigned)b1[i] << 16);
    }
    __syncthreads();

    v16bf af[2], bfr[4];
#pragma unroll
    for (int i = 0; i < 2; ++i) {
      const int m = wm * 32 + i * 16 + lo;
      af[i] = cat8(*reinterpret_cast<const v8bf*>(&sA[m][8 * hi]),
                   *reinterpret_cast<const v8bf*>(&sA[m][16 + 8 * hi]));
    }
#pragma unroll
    for (int j = 0; j < 4; ++j) {
      const int n = wn * 64 + j * 16 + lo;
      bfr[j] = cat8(*reinterpret_cast<const v8bf*>(&sB[n][8 * hi]),
                    *reinterpret_cast<const v8bf*>(&sB[n][16 + 8 * hi]));
    }
#pragma unroll
    for (int i = 0; i < 2; ++i)
#pragma unroll
      for (int j = 0; j < 4; ++j)
        acc[i][j] = __builtin_amdgcn_wmma_f32_16x16x32_bf16(
            false, af[i], false, bfr[j], (short)0, acc[i][j], false, false);
    __syncthreads();
  }

#pragma unroll
  for (int i = 0; i < 2; ++i)
#pragma unroll
    for (int j = 0; j < 4; ++j) {
      const int n = wn * 64 + j * 16 + lo;
      const int mb = wm * 32 + i * 16 + 8 * hi;
      v8bf o8;
#pragma unroll
      for (int r = 0; r < 8; ++r) o8[r] = (bf16)acc[i][j][r];
      *reinterpret_cast<v8bf*>(&kvT[(size_t)bh * (DH * DH) + n * DH + mb]) = o8;
    }
}

__global__ __launch_bounds__(256) void k_den(const bf16* __restrict__ pq,
                                             const float* __restrict__ z,
                                             float* __restrict__ den) {
  int i = blockIdx.x * 256 + threadIdx.x;
  int h = i & 7, sb = i >> 3;
  int b = sb & (B_SZ - 1);
  const bf16* p = pq + (size_t)sb * D_MODEL + h * DH;
  const float* zz = z + ((size_t)b * NH + h) * DH;
  float a = 0.f;
#pragma unroll 8
  for (int d = 0; d < DH; ++d) a += (float)p[d] * zz[d];
  den[i] = a;
}

// att[s,b,h,e] = (pq . kv)/(den+eps); one block per (s-block, b*h); K = DH.
__global__ __launch_bounds__(256) void k_num(const bf16* __restrict__ pq,
                                             const bf16* __restrict__ kvT,
                                             const float* __restrict__ den,
                                             bf16* __restrict__ att) {
  __shared__ bf16 sA[128][40];   // [s][d]
  __shared__ bf16 sB[128][40];   // [e][d]
  const int bh = blockIdx.y;
  const int b = bh >> 3, hh = bh & 7;
  const int sBlk = blockIdx.x * 128;
  const int t = threadIdx.x;
  const int wave = t >> 5, lane = t & 31;
  const int lo = lane & 15, hi = lane >> 4;
  const int wm = wave >> 1, wn = wave & 1;
  const size_t headOff = (size_t)b * D_MODEL + hh * DH;
  const size_t rowStride = (size_t)B_SZ * D_MODEL;
  const bf16* kvh = kvT + (size_t)bh * (DH * DH);

  const v8f vz = {0.f,0.f,0.f,0.f,0.f,0.f,0.f,0.f};
  v8f acc[2][4];
#pragma unroll
  for (int i = 0; i < 2; ++i)
#pragma unroll
    for (int j = 0; j < 4; ++j) acc[i][j] = vz;

  for (int k0 = 0; k0 < DH; k0 += 32) {
#pragma unroll
    for (int c = t; c < 512; c += 256) {
      const int row = c >> 2, part = c & 3;
      *reinterpret_cast<u32x4*>(&sA[row][part * 8]) =
          *reinterpret_cast<const u32x4*>(pq + (size_t)(sBlk + row) * rowStride +
                                          headOff + k0 + part * 8);
      *reinterpret_cast<u32x4*>(&sB[row][part * 8]) =
          *reinterpret_cast<const u32x4*>(kvh + (size_t)row * DH + k0 + part * 8);
    }
    __syncthreads();

    v16bf af[2], bfr[4];
#pragma unroll
    for (int i = 0; i < 2; ++i) {
      const int m = wm * 32 + i * 16 + lo;
      af[i] = cat8(*reinterpret_cast<const v8bf*>(&sA[m][8 * hi]),
                   *reinterpret_cast<const v8bf*>(&sA[m][16 + 8 * hi]));
    }
#pragma unroll
    for (int j = 0; j < 4; ++j) {
      const int n = wn * 64 + j * 16 + lo;
      bfr[j] = cat8(*reinterpret_cast<const v8bf*>(&sB[n][8 * hi]),
                    *reinterpret_cast<const v8bf*>(&sB[n][16 + 8 * hi]));
    }
#pragma unroll
    for (int i = 0; i < 2; ++i)
#pragma unroll
      for (int j = 0; j < 4; ++j)
        acc[i][j] = __builtin_amdgcn_wmma_f32_16x16x32_bf16(
            false, af[i], false, bfr[j], (short)0, acc[i][j], false, false);
    __syncthreads();
  }

#pragma unroll
  for (int i = 0; i < 2; ++i)
#pragma unroll
    for (int j = 0; j < 4; ++j) {
      const int n = wn * 64 + j * 16 + lo;
#pragma unroll
      for (int r = 0; r < 8; ++r) {
        const int m = sBlk + wm * 32 + i * 16 + 8 * hi + r;   // s index
        const float dn = den[((size_t)m * B_SZ + b) * NH + hh] + 1e-6f;
        att[((size_t)m * B_SZ + b) * D_MODEL + hh * DH + n] =
            (bf16)(acc[i][j][r] / dn);
      }
    }
}

extern "C" void kernel_launch(void* const* d_in, const int* in_sizes, int n_in,
                              void* d_out, int out_size, void* d_ws, size_t ws_size,
                              hipStream_t stream) {
  (void)in_sizes; (void)n_in; (void)out_size; (void)ws_size;
  const float* x  = (const float*)d_in[0];
  const float* Wq = (const float*)d_in[1];
  const float* bq = (const float*)d_in[2];
  const float* Wk = (const float*)d_in[3];
  const float* bk = (const float*)d_in[4];
  const float* Wv = (const float*)d_in[5];
  const float* bv = (const float*)d_in[6];
  const float* Wo = (const float*)d_in[7];
  const float* bo = (const float*)d_in[8];

  const size_t SB  = (size_t)S_LEN * B_SZ;
  const size_t SBD = SB * D_MODEL;
  char* ws = (char*)d_ws;
  bf16* hbf = (bf16*)ws;  ws += SBD * 2;                  // h; reused as att
  bf16* pq  = (bf16*)ws;  ws += SBD * 2;
  bf16* pk  = (bf16*)ws;  ws += SBD * 2;
  bf16* vb  = (bf16*)ws;  ws += SBD * 2;
  bf16* wqb = (bf16*)ws;  ws += (size_t)D_MODEL * D_MODEL * 2;
  bf16* wkb = (bf16*)ws;  ws += (size_t)D_MODEL * D_MODEL * 2;
  bf16* wvb = (bf16*)ws;  ws += (size_t)D_MODEL * D_MODEL * 2;
  bf16* wob = (bf16*)ws;  ws += (size_t)D_MODEL * D_MODEL * 2;
  bf16* kvT = (bf16*)ws;  ws += (size_t)B_SZ * NH * DH * DH * 2;
  float* z   = (float*)ws; ws += (size_t)B_SZ * D_MODEL * 4;
  float* den = (float*)ws; ws += SB * NH * 4;

  k_embed<<<(unsigned)(SBD / 256), 256, 0, stream>>>(x, hbf);
  const int wcnt = D_MODEL * D_MODEL;
  k_cvt<<<wcnt / 256, 256, 0, stream>>>(Wq, wqb, wcnt);
  k_cvt<<<wcnt / 256, 256, 0, stream>>>(Wk, wkb, wcnt);
  k_cvt<<<wcnt / 256, 256, 0, stream>>>(Wv, wvb, wcnt);
  k_cvt<<<wcnt / 256, 256, 0, stream>>>(Wo, wob, wcnt);

  dim3 gp((unsigned)(SB / 256), D_MODEL / 128);
  k_gemm_xwT<1><<<gp, 256, 0, stream>>>(hbf, wqb, bq, pq, (int)SB, D_MODEL, D_MODEL);
  k_gemm_xwT<1><<<gp, 256, 0, stream>>>(hbf, wkb, bk, pk, (int)SB, D_MODEL, D_MODEL);
  k_gemm_xwT<0><<<gp, 256, 0, stream>>>(hbf, wvb, bv, vb, (int)SB, D_MODEL, D_MODEL);

  k_zsum<<<(B_SZ * D_MODEL) / 256, 256, 0, stream>>>(pk, z);
  k_kv<<<B_SZ * NH, 256, 0, stream>>>(pk, vb, kvT);
  k_den<<<(unsigned)(SB * NH / 256), 256, 0, stream>>>(pq, z, den);

  dim3 gnum(S_LEN / 128, B_SZ * NH);
  k_num<<<gnum, 256, 0, stream>>>(pq, kvT, den, hbf);

  k_gemm_xwT<2><<<gp, 256, 0, stream>>>(hbf, wob, bo, d_out, (int)SB, D_MODEL, D_MODEL);
}
